// MultiHeadAttention_38208029065925
// MI455X (gfx1250) — compile-verified, matrix operand order checked by reading, not run on hardware
//
#include <hip/hip_runtime.h>
#include <hip/hip_bf16.h>
#include <cstdint>

// Problem constants (match the reference).
constexpr int kB = 4;
constexpr int kT = 2048;
constexpr int kC = 1024;
constexpr int kH = 16;
constexpr int kD = 64;

typedef __attribute__((ext_vector_type(16))) __bf16 v16bf;
typedef __attribute__((ext_vector_type(8)))  float  v8f;

// ---------------------------------------------------------------------------
// Fragment loader for CDNA5 16-bit WMMA operands (wave32).
// A-matrix 16x32 bf16 layout (ISA 7.12.2): lane owns row (lane&15); element j
// of the 16-wide vector holds K = 16*(j/8) + 8*(lane>>4) + (j%8).
// That is two contiguous groups of 8 bf16 (16 bytes each) per lane:
//   group0 at base + 8*half, group1 at base + 16 + 8*half.
// B-matrix uses the symmetric layout with lane owning column (lane&15), so the
// same loader works when `base` points at the (pre-transposed) column.
// ---------------------------------------------------------------------------
__device__ __forceinline__ v16bf load_frag16(const __bf16* base, int half) {
  union { v16bf v; uint4 u[2]; } f;
  f.u[0] = *reinterpret_cast<const uint4*>(base + (half << 3));
  f.u[1] = *reinterpret_cast<const uint4*>(base + 16 + (half << 3));
  return f.v;
}

// ---------------------------------------------------------------------------
// Elementwise fp32 -> bf16 conversion.
// ---------------------------------------------------------------------------
__global__ void cvt_bf16_kernel(const float* __restrict__ src,
                                __bf16* __restrict__ dst, int n) {
  int i = blockIdx.x * blockDim.x + threadIdx.x;
  if (i < n) dst[i] = (__bf16)src[i];
}

// [H,C,D] fp32 -> [H,D,C] bf16 (so GEMM B-columns are contiguous).
__global__ void transpose_w_kernel(const float* __restrict__ src,
                                   __bf16* __restrict__ dst) {
  int i = blockIdx.x * blockDim.x + threadIdx.x;  // H*C*D threads
  int h = i / (kC * kD);
  int rem = i % (kC * kD);
  int c = rem / kD;
  int d = rem % kD;
  dst[((size_t)h * kD + d) * kC + c] = (__bf16)src[i];
}

// ---------------------------------------------------------------------------
// QKV projection: out[b,h,t,d] = sum_c x[b,t,c] * W[h,c,d].
// One wave computes a 16(t) x 64(d) tile. WT is [H,D,C] bf16.
// transpose_out = 0 -> store [B,H,T,D]; = 1 -> store [B,H,D,T] (for V).
// ---------------------------------------------------------------------------
__global__ __launch_bounds__(32) void qkv_proj_kernel(
    const __bf16* __restrict__ xb,   // [B*T, C]
    const __bf16* __restrict__ WT,   // [H, D, C]
    __bf16* __restrict__ outp,
    int transpose_out) {
  const int mt   = blockIdx.x;            // 0..(B*T/16 - 1)
  const int h    = blockIdx.y;            // head
  const int lane = threadIdx.x;
  const int nr   = lane & 15;
  const int half = lane >> 4;

  const __bf16* arow = xb + (size_t)(mt * 16 + nr) * kC;

  v8f acc[4] = {};
  for (int kk = 0; kk < kC; kk += 32) {
    v16bf a = load_frag16(arow + kk, half);
#pragma unroll
    for (int nt = 0; nt < 4; ++nt) {
      const __bf16* brow = WT + ((size_t)h * kD + nt * 16 + nr) * kC + kk;
      v16bf b = load_frag16(brow, half);
      acc[nt] = __builtin_amdgcn_wmma_f32_16x16x32_bf16(
          false, a, false, b, (short)0, acc[nt], false, false);
    }
  }

  const int b_ = (mt * 16) / kT;   // 16-row tiles never straddle a batch
  const int t0 = (mt * 16) % kT;
#pragma unroll
  for (int nt = 0; nt < 4; ++nt) {
#pragma unroll
    for (int r = 0; r < 8; ++r) {
      const int trow = t0 + r + 8 * half;       // C/D layout: row r + 8*half
      const int d    = nt * 16 + nr;            // col = lane&15
      const __bf16 val = (__bf16)acc[nt][r];
      if (!transpose_out)
        outp[(((size_t)b_ * kH + h) * kT + trow) * kD + d] = val;
      else
        outp[(((size_t)b_ * kH + h) * kD + d) * kT + trow] = val;
    }
  }
}

// ---------------------------------------------------------------------------
// Flash attention (causal). One wave per (b, h, 16-row q tile).
// q,k: [B,H,T,D] bf16 ; vT: [B,H,D,T] bf16 ; attnb: [B,T,C] bf16 (concat heads)
// ---------------------------------------------------------------------------
__global__ __launch_bounds__(32) void attn_kernel(
    const __bf16* __restrict__ q,
    const __bf16* __restrict__ k,
    const __bf16* __restrict__ vT,
    __bf16* __restrict__ attnb) {
  __shared__ float sP[16][33];   // 16 q-rows x 32 chunk-cols (+pad)
  __shared__ float sAlpha[16];
  __shared__ float sL[16];

  const int tt   = blockIdx.x;   // q tile
  const int h    = blockIdx.y;
  const int b    = blockIdx.z;
  const int lane = threadIdx.x;
  const int nr   = lane & 15;
  const int half = lane >> 4;
  const int t0   = tt * 16;

  const __bf16* qbase = q  + (((size_t)b * kH + h) * kT) * kD;
  const __bf16* kbase = k  + (((size_t)b * kH + h) * kT) * kD;
  const __bf16* vbase = vT + (((size_t)b * kH + h) * kD) * kT;

  // Q A-fragments (lane owns row t0+nr), covering D=64 as two K=32 slabs.
  const v16bf qa0 = load_frag16(qbase + (size_t)(t0 + nr) * kD + 0,  half);
  const v16bf qa1 = load_frag16(qbase + (size_t)(t0 + nr) * kD + 32, half);

  v8f   o[4] = {};                      // O accum: col d = nt*16+nr, rows r+8*half
  float m_i = -__builtin_inff();
  float l_i = 0.f;
  const float scale = 0.03125f;         // 1024^-0.5

  const int qmax = t0 + 15;
  for (int s0 = 0; s0 <= qmax; s0 += 32) {
    if (s0 + 32 <= qmax) {              // prefetch next chunk's K and V rows
      __builtin_prefetch(kbase + (size_t)(s0 + 32 + nr) * kD, 0, 1);
      __builtin_prefetch(vbase + (size_t)nr * kT + s0 + 32, 0, 1);
    }
    // ---- S = Q * K^T for 32 key columns (two 16-col tiles) ----
#pragma unroll
    for (int j = 0; j < 2; ++j) {
      const __bf16* krow = kbase + (size_t)(s0 + j * 16 + nr) * kD;
      v16bf kb0 = load_frag16(krow + 0,  half);
      v16bf kb1 = load_frag16(krow + 32, half);
      v8f s = {};
      s = __builtin_amdgcn_wmma_f32_16x16x32_bf16(false, qa0, false, kb0,
                                                  (short)0, s, false, false);
      s = __builtin_amdgcn_wmma_f32_16x16x32_bf16(false, qa1, false, kb1,
                                                  (short)0, s, false, false);
      const int scol = s0 + j * 16 + nr;
#pragma unroll
      for (int r = 0; r < 8; ++r) {
        const int qrow = t0 + r + 8 * half;
        sP[r + 8 * half][j * 16 + nr] =
            (scol <= qrow) ? s[r] * scale : -__builtin_inff();
      }
    }
    __syncthreads();

    // ---- online softmax: lane handles q-row nr (halves duplicate) ----
    float rmax = -__builtin_inff();
#pragma unroll
    for (int c = 0; c < 32; ++c) rmax = fmaxf(rmax, sP[nr][c]);
    const float m_new = fmaxf(m_i, rmax);
    const float alpha = __expf(m_i - m_new);   // first chunk: exp(-inf)=0
    float rsum = 0.f;
    float e[32];
#pragma unroll
    for (int c = 0; c < 32; ++c) {
      e[c] = __expf(sP[nr][c] - m_new);
      rsum += e[c];
    }
    if (half == 0) {
#pragma unroll
      for (int c = 0; c < 32; ++c) sP[nr][c] = e[c];
      sAlpha[nr] = alpha;
    }
    m_i = m_new;
    l_i = l_i * alpha + rsum;
    __syncthreads();

    // ---- P A-fragment (row nr, K = chunk column per 16-bit A layout) ----
    v16bf pa;
#pragma unroll
    for (int j = 0; j < 16; ++j) {
      const int col = ((j >> 3) << 4) + (half << 3) + (j & 7);
      pa[j] = (__bf16)sP[nr][col];
    }
    float ar[8];
#pragma unroll
    for (int r = 0; r < 8; ++r) ar[r] = sAlpha[r + 8 * half];

    // ---- O = O*alpha + P * V  (4 WMMAs: d tiles of 16) ----
#pragma unroll
    for (int nt = 0; nt < 4; ++nt) {
#pragma unroll
      for (int r = 0; r < 8; ++r) o[nt][r] *= ar[r];
      const __bf16* vrow = vbase + (size_t)(nt * 16 + nr) * kT + s0;
      v16bf vb = load_frag16(vrow, half);
      o[nt] = __builtin_amdgcn_wmma_f32_16x16x32_bf16(
          false, pa, false, vb, (short)0, o[nt], false, false);
    }
    __syncthreads();   // sP/sAlpha reused next chunk
  }

  if (half == 0) sL[nr] = l_i;
  __syncthreads();

  // store normalized O, concat heads: attnb[b, t0+row, h*64 + d]
  __bf16* obase = attnb + ((size_t)b * kT + t0) * kC + h * kD;
#pragma unroll
  for (int nt = 0; nt < 4; ++nt) {
#pragma unroll
    for (int r = 0; r < 8; ++r) {
      const int row = r + 8 * half;
      const float val = o[nt][r] / sL[row];
      obase[(size_t)row * kC + nt * 16 + nr] = (__bf16)val;
    }
  }
}

// ---------------------------------------------------------------------------
// Output projection: out[m,n] = sum_k attn[m,k] * Wo[n,k] + bo[n]  (out=A@Wo^T)
// One wave: 16(m) x 64(n) tile. Wo rows are B columns -> contiguous.
// ---------------------------------------------------------------------------
__global__ __launch_bounds__(32) void out_proj_kernel(
    const __bf16* __restrict__ ab,    // [B*T, C]
    const __bf16* __restrict__ Wob,   // [C, C]
    const float*  __restrict__ bo,    // [C]
    float* __restrict__ out) {        // [B*T, C]
  const int mt   = blockIdx.x;        // 0..511
  const int nb   = blockIdx.y;        // 0..15 (64-wide col blocks)
  const int lane = threadIdx.x;
  const int nr   = lane & 15;
  const int half = lane >> 4;

  const __bf16* arow = ab + (size_t)(mt * 16 + nr) * kC;
  v8f acc[4] = {};
  for (int kk = 0; kk < kC; kk += 32) {
    v16bf a = load_frag16(arow + kk, half);
#pragma unroll
    for (int nt = 0; nt < 4; ++nt) {
      const __bf16* brow = Wob + (size_t)(nb * 64 + nt * 16 + nr) * kC + kk;
      v16bf bf = load_frag16(brow, half);
      acc[nt] = __builtin_amdgcn_wmma_f32_16x16x32_bf16(
          false, a, false, bf, (short)0, acc[nt], false, false);
    }
  }
#pragma unroll
  for (int nt = 0; nt < 4; ++nt) {
    const int n = nb * 64 + nt * 16 + nr;
    const float bias = bo[n];
#pragma unroll
    for (int r = 0; r < 8; ++r)
      out[(size_t)(mt * 16 + r + 8 * half) * kC + n] = acc[nt][r] + bias;
  }
}

// ---------------------------------------------------------------------------
// Host launcher.
// ---------------------------------------------------------------------------
extern "C" void kernel_launch(void* const* d_in, const int* in_sizes, int n_in,
                              void* d_out, int out_size, void* d_ws,
                              size_t ws_size, hipStream_t stream) {
  const float* x  = (const float*)d_in[0];
  const float* Wq = (const float*)d_in[1];
  const float* Wk = (const float*)d_in[2];
  const float* Wv = (const float*)d_in[3];
  const float* Wo = (const float*)d_in[4];
  const float* bo = (const float*)d_in[5];
  float* out = (float*)d_out;

  char* ws = (char*)d_ws;
  size_t off = 0;
  auto alloc = [&](size_t bytes) -> char* {
    char* p = ws + off;
    off += (bytes + 255) & ~(size_t)255;
    return p;
  };

  const size_t nX  = (size_t)kB * kT * kC;     // 8.4M
  const size_t nW  = (size_t)kH * kC * kD;     // 1.0M per projection weight
  const size_t nWo = (size_t)kC * kC;
  const size_t nQ  = (size_t)kB * kH * kT * kD;

  __bf16* xb  = (__bf16*)alloc(nX * 2);
  __bf16* WqT = (__bf16*)alloc(nW * 2);
  __bf16* WkT = (__bf16*)alloc(nW * 2);
  __bf16* WvT = (__bf16*)alloc(nW * 2);
  __bf16* Wob = (__bf16*)alloc(nWo * 2);
  __bf16* qb  = (__bf16*)alloc(nQ * 2);
  __bf16* kb  = (__bf16*)alloc(nQ * 2);
  __bf16* vTb = (__bf16*)alloc(nQ * 2);
  __bf16* abf = (__bf16*)alloc(nX * 2);

  // Stage 0: precision conversion / weight transposition.
  cvt_bf16_kernel<<<(int)((nX + 255) / 256), 256, 0, stream>>>(x, xb, (int)nX);
  cvt_bf16_kernel<<<(int)((nWo + 255) / 256), 256, 0, stream>>>(Wo, Wob, (int)nWo);
  transpose_w_kernel<<<(int)((nW + 255) / 256), 256, 0, stream>>>(Wq, WqT);
  transpose_w_kernel<<<(int)((nW + 255) / 256), 256, 0, stream>>>(Wk, WkT);
  transpose_w_kernel<<<(int)((nW + 255) / 256), 256, 0, stream>>>(Wv, WvT);

  // Stage 1: Q/K/V projections (V stored transposed for the P*V GEMM).
  const dim3 gProj(kB * kT / 16, kH);
  qkv_proj_kernel<<<gProj, 32, 0, stream>>>(xb, WqT, qb, 0);
  qkv_proj_kernel<<<gProj, 32, 0, stream>>>(xb, WkT, kb, 0);
  qkv_proj_kernel<<<gProj, 32, 0, stream>>>(xb, WvT, vTb, 1);

  // Stage 2: causal flash attention.
  attn_kernel<<<dim3(kT / 16, kH, kB), 32, 0, stream>>>(qb, kb, vTb, abf);

  // Stage 3: output projection + bias.
  out_proj_kernel<<<dim3(kB * kT / 16, kC / 64), 32, 0, stream>>>(abf, Wob, bo,
                                                                  out);
}
